// CQAttention_90452011253974
// MI455X (gfx1250) — compile-verified, matrix operand order checked by reading, not run on hardware
//
#include <hip/hip_runtime.h>
#include <hip/hip_bf16.h>

typedef __bf16 bf16_t;
typedef __attribute__((ext_vector_type(16))) __bf16 v16bf;
typedef __attribute__((ext_vector_type(8)))  __bf16 v8bf;
typedef __attribute__((ext_vector_type(8)))  float  v8f;
typedef int v4i_vs __attribute__((vector_size(16)));
typedef __attribute__((address_space(1))) v4i_vs* gptr_b128;
typedef __attribute__((address_space(3))) v4i_vs* lptr_b128;

#define BB 32
#define DD 128
#define LC 2048
#define LQ 256
#define KSPLIT 8
#define NEG_INF_F (-1e30f)

#if __has_builtin(__builtin_amdgcn_global_load_async_to_lds_b128)
#define HAVE_ASYNC_LDS 1
#else
#define HAVE_ASYNC_LDS 0
#endif

// 16-byte-per-lane staging copy: global -> LDS (async if available)
__device__ __forceinline__ void stage_copy16(const bf16_t* g, bf16_t* l) {
#if HAVE_ASYNC_LDS
    __builtin_amdgcn_global_load_async_to_lds_b128((gptr_b128)(g), (lptr_b128)(l), 0, 0);
#else
    uint4 v = *(const uint4*)g;
    *(uint4*)l = v;
#endif
}

__device__ __forceinline__ void stage_wait_le1() {
#if HAVE_ASYNC_LDS
#if __has_builtin(__builtin_amdgcn_s_wait_asynccnt)
    __builtin_amdgcn_s_wait_asynccnt(1);
#else
    asm volatile("s_wait_asynccnt 0x1" ::: "memory");
#endif
#endif
}
__device__ __forceinline__ void stage_wait_0() {
#if HAVE_ASYNC_LDS
#if __has_builtin(__builtin_amdgcn_s_wait_asynccnt)
    __builtin_amdgcn_s_wait_asynccnt(0);
#else
    asm volatile("s_wait_asynccnt 0x0" ::: "memory");
#endif
#endif
}

// 32-byte fragment from LDS via two 16B-aligned halves (row stride is 80B)
__device__ __forceinline__ v16bf load_lds_v16(const bf16_t* p) {
    union { v16bf v; v8bf h[2]; } u;
    u.h[0] = *(const v8bf*)(p);
    u.h[1] = *(const v8bf*)(p + 8);
    return u.v;
}

// ---------------------------------------------------------------------------
// 1) elementwise f32 -> bf16 convert
// ---------------------------------------------------------------------------
__global__ void cq_cvt_bf16(const float* __restrict__ in, bf16_t* __restrict__ out, int n) {
    int i = blockIdx.x * blockDim.x + threadIdx.x;
    if (i < n) out[i] = (bf16_t)in[i];
}

// ---------------------------------------------------------------------------
// 2) [B][D][L] f32 -> [B][L][D] bf16 transpose, optional per-d scale (w4mlu)
//    grid (L/32, D/32, B), block (32, 8)
// ---------------------------------------------------------------------------
__global__ void cq_transpose_scale(const float* __restrict__ in, bf16_t* __restrict__ out,
                                   const float* __restrict__ scale, int useScale, int L) {
    __shared__ float tile[32][33];
    int b = blockIdx.z;
    int i0 = blockIdx.x * 32, d0 = blockIdx.y * 32;
    const float* src = in + (size_t)b * DD * L;
    for (int r = threadIdx.y; r < 32; r += 8) {
        float v = src[(size_t)(d0 + r) * L + i0 + threadIdx.x];
        if (useScale) v *= scale[d0 + r];
        tile[r][threadIdx.x] = v;
    }
    __syncthreads();
    bf16_t* dst = out + (size_t)b * L * DD;
    for (int r = threadIdx.y; r < 32; r += 8) {
        dst[(size_t)(i0 + r) * DD + d0 + threadIdx.x] = (bf16_t)tile[threadIdx.x][r];
    }
}

// ---------------------------------------------------------------------------
// 3) per-token dot with weight vector: out[b][i] = sum_d in[b][d][i] * w[d]
// ---------------------------------------------------------------------------
__global__ void cq_dotw(const float* __restrict__ in, const float* __restrict__ w,
                        float* __restrict__ out, int L) {
    int b = blockIdx.y;
    int i = blockIdx.x * 256 + threadIdx.x;
    const float* src = in + (size_t)b * DD * L + i;
    float acc = 0.f;
    #pragma unroll 8
    for (int d = 0; d < DD; ++d) acc += src[(size_t)d * L] * w[d];
    out[(size_t)b * L + i] = acc;
}

// ---------------------------------------------------------------------------
// 4) S[b][i][j] = Ct[i]·Qs[j] + aC[i] + bQ[j] + bias   (WMMA bf16, K = 128)
//    grid (LQ/64, LC/64, B), block 128 (4 waves, each a 16-row strip, 4 j-tiles)
// ---------------------------------------------------------------------------
__global__ void cq_s_gemm(const bf16_t* __restrict__ Ct, const bf16_t* __restrict__ Qs,
                          const float* __restrict__ aC, const float* __restrict__ bQ,
                          const float* __restrict__ bias, float* __restrict__ S) {
    int b = blockIdx.z;
    int lane = threadIdx.x & 31;
    int wave = threadIdx.x >> 5;
    int i0 = blockIdx.y * 64 + wave * 16;
    int j0 = blockIdx.x * 64;
    int mrow = lane & 15;
    int khalf = (lane >> 4) * 16;
    int rbase = (lane >> 4) * 8;

    const bf16_t* Arow = Ct + (size_t)b * LC * DD + (size_t)(i0 + mrow) * DD + khalf;
    const bf16_t* Brow = Qs + (size_t)b * LQ * DD;
    v16bf a[4];
    #pragma unroll
    for (int k = 0; k < 4; ++k) a[k] = *(const v16bf*)(Arow + k * 32);

    float biasv = bias[0];
    float* Sout = S + (size_t)b * LC * LQ;
    const float* aCb = aC + (size_t)b * LC;
    const float* bQb = bQ + (size_t)b * LQ;

    for (int jt = 0; jt < 4; ++jt) {
        int j = j0 + jt * 16;
        v8f acc = {};
        #pragma unroll
        for (int k = 0; k < 4; ++k) {
            v16bf bm = *(const v16bf*)(Brow + (size_t)(j + mrow) * DD + k * 32 + khalf);
            acc = __builtin_amdgcn_wmma_f32_16x16x32_bf16(false, a[k], false, bm,
                                                          (short)0, acc, false, false);
        }
        float bq = bQb[j + mrow];
        #pragma unroll
        for (int v = 0; v < 8; ++v) {
            int ii = i0 + rbase + v;
            Sout[(size_t)ii * LQ + j + mrow] = acc[v] + aCb[ii] + bq + biasv;
        }
    }
}

// ---------------------------------------------------------------------------
// 5) row softmax over Lq (Qmask) -> S1 bf16.  grid (LC/8, B), block 256
// ---------------------------------------------------------------------------
__global__ void cq_row_softmax(const float* __restrict__ S, const int* __restrict__ Qmask,
                               bf16_t* __restrict__ S1) {
    int b = blockIdx.y;
    int wave = threadIdx.x >> 5, lane = threadIdx.x & 31;
    int i = blockIdx.x * 8 + wave;
    const float* row = S + ((size_t)b * LC + i) * LQ;
    const int* qm = Qmask + (size_t)b * LQ;
    float x[8];
    float mx = -3.4e38f;
    #pragma unroll
    for (int k = 0; k < 8; ++k) {
        int j = k * 32 + lane;
        float v = row[j];
        if (qm[j] == 0) v += NEG_INF_F;
        x[k] = v;
        mx = fmaxf(mx, v);
    }
    #pragma unroll
    for (int off = 16; off >= 1; off >>= 1) mx = fmaxf(mx, __shfl_xor(mx, off, 32));
    float s = 0.f;
    #pragma unroll
    for (int k = 0; k < 8; ++k) { x[k] = __expf(x[k] - mx); s += x[k]; }
    #pragma unroll
    for (int off = 16; off >= 1; off >>= 1) s += __shfl_xor(s, off, 32);
    float inv = 1.0f / s;
    bf16_t* o = S1 + ((size_t)b * LC + i) * LQ;
    #pragma unroll
    for (int k = 0; k < 8; ++k) o[k * 32 + lane] = (bf16_t)(x[k] * inv);
}

// ---------------------------------------------------------------------------
// 6) column softmax over Lc (Cmask), stored transposed: S2T[b][j][i] bf16
//    grid (LQ/32, B), block (32, 8)
// ---------------------------------------------------------------------------
__global__ void cq_col_softmax(const float* __restrict__ S, const int* __restrict__ Cmask,
                               bf16_t* __restrict__ S2T) {
    __shared__ float red[8][32];
    __shared__ float cmax[32];
    __shared__ float cinv[32];
    __shared__ float tile[32][33];
    int b = blockIdx.y;
    int j0 = blockIdx.x * 32;
    int x = threadIdx.x, y = threadIdx.y;
    const float* Sb = S + (size_t)b * LC * LQ;
    const int* cm = Cmask + (size_t)b * LC;

    float mx = -3.4e38f;
    for (int i = y; i < LC; i += 8) {
        float v = Sb[(size_t)i * LQ + j0 + x];
        if (cm[i] == 0) v += NEG_INF_F;
        mx = fmaxf(mx, v);
    }
    red[y][x] = mx;
    __syncthreads();
    if (y == 0) {
        #pragma unroll
        for (int k = 1; k < 8; ++k) mx = fmaxf(mx, red[k][x]);
        cmax[x] = mx;
    }
    __syncthreads();
    mx = cmax[x];

    float s = 0.f;
    for (int i = y; i < LC; i += 8) {
        float v = Sb[(size_t)i * LQ + j0 + x];
        if (cm[i] == 0) v += NEG_INF_F;
        s += __expf(v - mx);
    }
    __syncthreads();
    red[y][x] = s;
    __syncthreads();
    if (y == 0) {
        #pragma unroll
        for (int k = 1; k < 8; ++k) s += red[k][x];
        cinv[x] = 1.0f / s;
    }
    __syncthreads();
    float inv = cinv[x];

    for (int i0 = 0; i0 < LC; i0 += 32) {
        #pragma unroll
        for (int k = 0; k < 4; ++k) {
            int il = y + 8 * k;
            int i = i0 + il;
            float v = Sb[(size_t)i * LQ + j0 + x];
            if (cm[i] == 0) v += NEG_INF_F;
            tile[il][x] = __expf(v - mx) * inv;
        }
        __syncthreads();
        #pragma unroll
        for (int k = 0; k < 4; ++k) {
            int jl = y + 8 * k;
            S2T[((size_t)b * LQ + j0 + jl) * LC + i0 + x] = (bf16_t)tile[x][jl];
        }
        __syncthreads();
    }
}

// ---------------------------------------------------------------------------
// 7) VTpart[s][b][d][j] = sum_{i in seg s} Cb[b][d][i] * S2T[b][j][i]
//    K-split x8 for occupancy.  grid (LQ/32, KSPLIT, B), block 128:
//    4 waves x (32d x 32j) register tiles; shared 32x32 B-tile double-buffered
//    in LDS via async global->LDS copies (s_wait_asynccnt pipelining).
// ---------------------------------------------------------------------------
__global__ void cq_vt_gemm(const bf16_t* __restrict__ Cb, const bf16_t* __restrict__ S2T,
                           float* __restrict__ VTpart) {
    __shared__ bf16_t sB[2][32][40];   // 80B row stride: 20-bank step, b128-aligned
    int b = blockIdx.z;
    int j0 = blockIdx.x * 32;
    int kbase = blockIdx.y * (LC / KSPLIT);
    int t = threadIdx.x;
    int lane = t & 31, wave = t >> 5;
    int d0 = wave * 32;
    int mrow = lane & 15, khalf = (lane >> 4) * 16, rbase = (lane >> 4) * 8;

    // staging mapping: thread t -> row t>>2 (j), 16B segment (t&3)
    const bf16_t* gsrc = S2T + (size_t)b * LQ * LC + (size_t)(j0 + (t >> 2)) * LC
                       + kbase + (t & 3) * 8;
    bf16_t* ldst[2] = { &sB[0][t >> 2][(t & 3) * 8], &sB[1][t >> 2][(t & 3) * 8] };

    const bf16_t* A0 = Cb + (size_t)b * DD * LC + (size_t)(d0 + mrow) * LC + kbase + khalf;
    const bf16_t* A1 = A0 + (size_t)16 * LC;

    v8f acc00 = {}, acc01 = {}, acc10 = {}, acc11 = {};

    stage_copy16(gsrc, ldst[0]);
    #pragma unroll
    for (int kc = 0; kc < (LC / KSPLIT) / 32; ++kc) {   // 8 stages
        int cur = kc & 1;
        if (kc + 1 < (LC / KSPLIT) / 32) {
            stage_copy16(gsrc + (kc + 1) * 32, ldst[cur ^ 1]);
            stage_wait_le1();
        } else {
            stage_wait_0();
        }
        __syncthreads();
        v16bf a0 = *(const v16bf*)(A0 + kc * 32);
        v16bf a1 = *(const v16bf*)(A1 + kc * 32);
        v16bf b0 = load_lds_v16(&sB[cur][mrow][khalf]);
        v16bf b1 = load_lds_v16(&sB[cur][16 + mrow][khalf]);
        acc00 = __builtin_amdgcn_wmma_f32_16x16x32_bf16(false, a0, false, b0, (short)0, acc00, false, false);
        acc01 = __builtin_amdgcn_wmma_f32_16x16x32_bf16(false, a0, false, b1, (short)0, acc01, false, false);
        acc10 = __builtin_amdgcn_wmma_f32_16x16x32_bf16(false, a1, false, b0, (short)0, acc10, false, false);
        acc11 = __builtin_amdgcn_wmma_f32_16x16x32_bf16(false, a1, false, b1, (short)0, acc11, false, false);
        __syncthreads();
    }

    float* pb = VTpart + ((size_t)blockIdx.y * BB + b) * (size_t)DD * LQ;
    #pragma unroll
    for (int v = 0; v < 8; ++v) {
        int dA = d0 + rbase + v, dB = d0 + 16 + rbase + v;
        pb[(size_t)dA * LQ + j0 + mrow]      = acc00[v];
        pb[(size_t)dA * LQ + j0 + 16 + mrow] = acc01[v];
        pb[(size_t)dB * LQ + j0 + mrow]      = acc10[v];
        pb[(size_t)dB * LQ + j0 + 16 + mrow] = acc11[v];
    }
}

// ---------------------------------------------------------------------------
// 7b) reduce K-split partials -> VT bf16 (deterministic, no atomics)
// ---------------------------------------------------------------------------
__global__ void cq_vt_reduce(const float* __restrict__ part, bf16_t* __restrict__ VT) {
    int i = blockIdx.x * 256 + threadIdx.x;
    const size_t n = (size_t)BB * DD * LQ;
    float s = 0.f;
    #pragma unroll
    for (int k = 0; k < KSPLIT; ++k) s += part[(size_t)k * n + i];
    VT[i] = (bf16_t)s;
}

// ---------------------------------------------------------------------------
// 8) final: A^T = Q·S1^T, Bt^T = VT·S1^T (K = 256); 2x-in-i register tiling.
//    grid (LC/128, D/16, B), block 128
// ---------------------------------------------------------------------------
__global__ void cq_final(const bf16_t* __restrict__ Qb, const bf16_t* __restrict__ VT,
                         const bf16_t* __restrict__ S1, const float* __restrict__ Cf,
                         float* __restrict__ out) {
    int b = blockIdx.z;
    int lane = threadIdx.x & 31, wave = threadIdx.x >> 5;
    int d0 = blockIdx.y * 16;
    int i0 = blockIdx.x * 128 + wave * 32;
    int mrow = lane & 15, khalf = (lane >> 4) * 16, rbase = (lane >> 4) * 8;
    const bf16_t* Aq  = Qb + (size_t)b * DD * LQ + (size_t)(d0 + mrow) * LQ + khalf;
    const bf16_t* Av  = VT + (size_t)b * DD * LQ + (size_t)(d0 + mrow) * LQ + khalf;
    const bf16_t* Bs0 = S1 + (size_t)b * LC * LQ + (size_t)(i0 + mrow) * LQ + khalf;
    const bf16_t* Bs1 = Bs0 + (size_t)16 * LQ;
    v8f aA0 = {}, aA1 = {}, aB0 = {}, aB1 = {};
    #pragma unroll
    for (int k = 0; k < LQ; k += 32) {
        v16bf q  = *(const v16bf*)(Aq + k);
        v16bf w  = *(const v16bf*)(Av + k);
        v16bf s0 = *(const v16bf*)(Bs0 + k);
        v16bf s1 = *(const v16bf*)(Bs1 + k);
        aA0 = __builtin_amdgcn_wmma_f32_16x16x32_bf16(false, q, false, s0, (short)0, aA0, false, false);
        aA1 = __builtin_amdgcn_wmma_f32_16x16x32_bf16(false, q, false, s1, (short)0, aA1, false, false);
        aB0 = __builtin_amdgcn_wmma_f32_16x16x32_bf16(false, w, false, s0, (short)0, aB0, false, false);
        aB1 = __builtin_amdgcn_wmma_f32_16x16x32_bf16(false, w, false, s1, (short)0, aB1, false, false);
    }
    const float* Cbb = Cf + (size_t)b * DD * LC;
    float* ob = out + (size_t)b * 4 * DD * LC;
    #pragma unroll
    for (int v = 0; v < 8; ++v) {
        int d = d0 + rbase + v;
        int ia = i0 + mrow, ib = i0 + 16 + mrow;
        float cva = Cbb[(size_t)d * LC + ia];
        float cvb = Cbb[(size_t)d * LC + ib];
        float a0 = aA0[v], a1 = aA1[v], b0 = aB0[v], b1 = aB1[v];
        ob[(size_t)(0 * DD + d) * LC + ia] = cva;
        ob[(size_t)(0 * DD + d) * LC + ib] = cvb;
        ob[(size_t)(1 * DD + d) * LC + ia] = a0;
        ob[(size_t)(1 * DD + d) * LC + ib] = a1;
        ob[(size_t)(2 * DD + d) * LC + ia] = cva * a0;
        ob[(size_t)(2 * DD + d) * LC + ib] = cvb * a1;
        ob[(size_t)(3 * DD + d) * LC + ia] = cva * b0;
        ob[(size_t)(3 * DD + d) * LC + ib] = cvb * b1;
    }
}

// ---------------------------------------------------------------------------
extern "C" void kernel_launch(void* const* d_in, const int* in_sizes, int n_in,
                              void* d_out, int out_size, void* d_ws, size_t ws_size,
                              hipStream_t stream) {
    const float* C     = (const float*)d_in[0];   // [B, D, Lc]
    const float* Q     = (const float*)d_in[1];   // [B, D, Lq]
    const int*   Cmask = (const int*)d_in[2];     // [B, Lc]
    const int*   Qmask = (const int*)d_in[3];     // [B, Lq]
    const float* w4C   = (const float*)d_in[4];   // [D]
    const float* w4Q   = (const float*)d_in[5];   // [D]
    const float* w4mlu = (const float*)d_in[6];   // [D]
    const float* bias  = (const float*)d_in[7];   // [1]
    float* out = (float*)d_out;                   // [B, 4D, Lc]

    size_t off = 0;
    auto carve = [&](size_t bytes) -> void* {
        void* p = (char*)d_ws + off;
        off = (off + bytes + 255) & ~(size_t)255;
        return p;
    };
    bf16_t* C_bf   = (bf16_t*)carve((size_t)BB * DD * LC * 2);          // [B][D][Lc]
    bf16_t* Ct_bf  = (bf16_t*)carve((size_t)BB * LC * DD * 2);          // [B][Lc][D]
    bf16_t* Q_bf   = (bf16_t*)carve((size_t)BB * DD * LQ * 2);          // [B][D][Lq]
    bf16_t* Qs_bf  = (bf16_t*)carve((size_t)BB * LQ * DD * 2);          // [B][Lq][D]
    float*  aC     = (float*)carve((size_t)BB * LC * 4);                // [B][Lc]
    float*  bQ     = (float*)carve((size_t)BB * LQ * 4);                // [B][Lq]
    float*  S      = (float*)carve((size_t)BB * LC * LQ * 4);           // [B][Lc][Lq]
    bf16_t* S1     = (bf16_t*)carve((size_t)BB * LC * LQ * 2);          // [B][Lc][Lq]
    bf16_t* S2T    = (bf16_t*)carve((size_t)BB * LQ * LC * 2);          // [B][Lq][Lc]
    bf16_t* VT     = (bf16_t*)carve((size_t)BB * DD * LQ * 2);          // [B][D][Lq]
    float*  VTpart = (float*)carve((size_t)KSPLIT * BB * DD * LQ * 4);  // [K][B][D][Lq]
    (void)ws_size; (void)in_sizes; (void)n_in; (void)out_size;

    {
        int nC = BB * DD * LC;
        cq_cvt_bf16<<<dim3((nC + 255) / 256), dim3(256), 0, stream>>>(C, C_bf, nC);
        int nQ = BB * DD * LQ;
        cq_cvt_bf16<<<dim3((nQ + 255) / 256), dim3(256), 0, stream>>>(Q, Q_bf, nQ);
    }
    cq_transpose_scale<<<dim3(LC / 32, DD / 32, BB), dim3(32, 8), 0, stream>>>(
        C, Ct_bf, nullptr, 0, LC);
    cq_transpose_scale<<<dim3(LQ / 32, DD / 32, BB), dim3(32, 8), 0, stream>>>(
        Q, Qs_bf, w4mlu, 1, LQ);
    cq_dotw<<<dim3(LC / 256, BB), dim3(256), 0, stream>>>(C, w4C, aC, LC);
    cq_dotw<<<dim3(LQ / 256, BB), dim3(256), 0, stream>>>(Q, w4Q, bQ, LQ);
    cq_s_gemm<<<dim3(LQ / 64, LC / 64, BB), dim3(128), 0, stream>>>(
        Ct_bf, Qs_bf, aC, bQ, bias, S);
    cq_row_softmax<<<dim3(LC / 8, BB), dim3(256), 0, stream>>>(S, Qmask, S1);
    cq_col_softmax<<<dim3(LQ / 32, BB), dim3(32, 8), 0, stream>>>(S, Cmask, S2T);
    cq_vt_gemm<<<dim3(LQ / 32, KSPLIT, BB), dim3(128), 0, stream>>>(C_bf, S2T, VTpart);
    cq_vt_reduce<<<dim3((BB * DD * LQ) / 256), dim3(256), 0, stream>>>(VTpart, VT);
    cq_final<<<dim3(LC / 128, DD / 16, BB), dim3(128), 0, stream>>>(Q_bf, VT, S1, C, out);
}
